// GCN_20452634263695
// MI455X (gfx1250) — matrix operand order repaired and checked by reading, not compiled, not run on hardware
//
#include <hip/hip_runtime.h>

// GCN 2-layer forward for MI455X (gfx1250, wave32, WMMA).
// GEMMs on v_wmma_f32_16x16x32_f16 (fp16 in, fp32 accumulate);
// edge aggregation via global_atomic_add_f32 (memory/atomic-bound part).

typedef __attribute__((ext_vector_type(16))) _Float16 v16h;
typedef __attribute__((ext_vector_type(8)))  _Float16 v8h;
typedef __attribute__((ext_vector_type(8)))  float    v8f;

#define C_IN      128
#define C_HID     128
#define C_OUT     40
#define C_OUT_PAD 48

// ---------------- elementwise conversion ----------------

__global__ void cvt_f32_to_f16(const float* __restrict__ in,
                               _Float16* __restrict__ out, int n) {
  int i = blockIdx.x * blockDim.x + threadIdx.x;
  if (i < n) out[i] = (_Float16)in[i];
}

// W2 (128x40 f32) -> zero-padded 128x48 f16 so WMMA N-tiles are clean.
__global__ void pad_w2_f16(const float* __restrict__ W2,
                           _Float16* __restrict__ W2h) {
  int i = blockIdx.x * blockDim.x + threadIdx.x;
  if (i < C_HID * C_OUT_PAD) {
    int r = i / C_OUT_PAD;
    int c = i - r * C_OUT_PAD;
    W2h[i] = (c < C_OUT) ? (_Float16)W2[r * C_OUT + c] : (_Float16)0.0f;
  }
}

// ---------------- degree / normalization ----------------

__global__ void degree_kernel(const int* __restrict__ dst,
                              float* __restrict__ deg, int E) {
  int e = blockIdx.x * blockDim.x + threadIdx.x;
  if (e < E) atomicAdd(&deg[dst[e]], 1.0f);
}

__global__ void dinv_kernel(const float* __restrict__ deg,
                            float* __restrict__ dinv, int n) {
  int i = blockIdx.x * blockDim.x + threadIdx.x;
  if (i < n) dinv[i] = rsqrtf(deg[i] + 1.0f);   // deg includes self-loop
}

// ---------------- WMMA GEMM: C[M x NT*16] = A[M x 128] * B[128 x NT*16] ----
// One wave per 16-row tile. K=128 in 4 steps of 32. NT accumulator tiles.
// A is row-major f16 (ld=128), B row-major f16 (ld=NT*16), C f32 (ld=NT*16).
// Requires gridDim.x == M/16 (M=50000 = 3125*16, exact).

template<int NT>
__global__ void wmma_gemm_k128(const _Float16* __restrict__ A,
                               const _Float16* __restrict__ B,
                               float* __restrict__ C) {
  const int lane = threadIdx.x & 31;
  const int lm   = lane & 15;
  const int half = lane >> 4;           // 0: lanes 0-15, 1: lanes 16-31
  const int r0   = blockIdx.x * 16;

  v8f zero = {};
  v8f acc[NT];
#pragma unroll
  for (int nt = 0; nt < NT; ++nt) acc[nt] = zero;

  // A fragment rows: lane (mod 16) owns matrix row r0+lm.
  const _Float16* Arow = A + (size_t)(r0 + lm) * C_IN;

#pragma unroll
  for (int ks = 0; ks < 128; ks += 32) {
    // ISA 16-bit A 16x32 layout: lanes 0-15 hold K = ks+[0..7] and ks+[16..23],
    // lanes 16-31 hold K = ks+[8..15] and ks+[24..31].
    const int kb = ks + half * 8;
    v8h alo = *(const v8h*)(Arow + kb);        // 16B aligned
    v8h ahi = *(const v8h*)(Arow + kb + 16);   // 16B aligned
    v16h a;
#pragma unroll
    for (int i = 0; i < 8; ++i) { a[i] = alo[i]; a[8 + i] = ahi[i]; }

    // B layout: lane holds column N = nt*16 + lm, K in the same order as A.
    const _Float16* Brow = B + (size_t)kb * (NT * 16) + lm;
#pragma unroll
    for (int nt = 0; nt < NT; ++nt) {
      v16h b;
#pragma unroll
      for (int i = 0; i < 8; ++i) {
        b[i]     = Brow[(size_t)i * (NT * 16) + nt * 16];
        b[8 + i] = Brow[(size_t)(16 + i) * (NT * 16) + nt * 16];
      }
      acc[nt] = __builtin_amdgcn_wmma_f32_16x16x32_f16(
          /*neg_a=*/false, a, /*neg_b=*/false, b,
          /*c_mod=*/(short)0, acc[nt], /*reuse_a=*/false, /*reuse_b=*/false);
    }
  }

  // C/D f32 layout: VGPR r -> row r0 + half*8 + r, col = nt*16 + lm.
  const int crow = r0 + half * 8;
#pragma unroll
  for (int nt = 0; nt < NT; ++nt) {
#pragma unroll
    for (int r = 0; r < 8; ++r)
      C[(size_t)(crow + r) * (NT * 16) + nt * 16 + lm] = acc[nt][r];
  }
}

// ---------------- edge scatter (layer 1: 128 channels) ----------------
// One wave per edge; each lane handles 4 contiguous channels (float4 gather).

__global__ void scatter_edges128(const int* __restrict__ src,
                                 const int* __restrict__ dst,
                                 const float* __restrict__ dinv,
                                 const float* __restrict__ h,     // [N,128]
                                 float* __restrict__ agg,         // [N,128]
                                 int E) {
  int wid  = (blockIdx.x * blockDim.x + threadIdx.x) >> 5;
  int lane = threadIdx.x & 31;
  if (wid >= E) return;
  int s = src[wid], d = dst[wid];
  float norm = dinv[s] * dinv[d];
  const float4 v = *(const float4*)(h + (size_t)s * C_HID + lane * 4);
  float* ap = agg + (size_t)d * C_HID + lane * 4;
  atomicAdd(ap + 0, v.x * norm);
  atomicAdd(ap + 1, v.y * norm);
  atomicAdd(ap + 2, v.z * norm);
  atomicAdd(ap + 3, v.w * norm);
}

// ---------------- self-loop + bias + ReLU -> f16 activations ----------

__global__ void selfloop_relu_f16(const float* __restrict__ agg,
                                  const float* __restrict__ h,
                                  const float* __restrict__ dinv,
                                  const float* __restrict__ bias,
                                  _Float16* __restrict__ out_f16, int n) {
  int i = blockIdx.x * blockDim.x + threadIdx.x;   // n = N*128
  if (i < n) {
    int row = i >> 7;            // /128
    int c   = i & 127;
    float di = dinv[row];
    float v  = agg[i] + h[i] * di * di + bias[c];
    out_f16[i] = (_Float16)(v > 0.0f ? v : 0.0f);
  }
}

// ---------------- edge scatter (layer 2: 40 channels, padded src ld=48) ---

__global__ void scatter_edges40(const int* __restrict__ src,
                                const int* __restrict__ dst,
                                const float* __restrict__ dinv,
                                const float* __restrict__ h2,  // [N,48] padded
                                float* __restrict__ out,       // [N,40]
                                int E) {
  int wid  = (blockIdx.x * blockDim.x + threadIdx.x) >> 5;
  int lane = threadIdx.x & 31;
  if (wid >= E) return;
  int s = src[wid], d = dst[wid];
  float norm = dinv[s] * dinv[d];
  float v0 = h2[(size_t)s * C_OUT_PAD + lane];
  atomicAdd(&out[(size_t)d * C_OUT + lane], v0 * norm);
  if (lane < C_OUT - 32) {
    float v1 = h2[(size_t)s * C_OUT_PAD + 32 + lane];
    atomicAdd(&out[(size_t)d * C_OUT + 32 + lane], v1 * norm);
  }
}

// ---------------- final: out += h2*dinv^2 + b2 ----------------

__global__ void final_add(const float* __restrict__ h2,   // [N,48] padded
                          const float* __restrict__ dinv,
                          const float* __restrict__ b2,
                          float* __restrict__ out, int n) { // n = N*40
  int i = blockIdx.x * blockDim.x + threadIdx.x;
  if (i < n) {
    int row = i / C_OUT;
    int c   = i - row * C_OUT;
    float di = dinv[row];
    out[i] += h2[(size_t)row * C_OUT_PAD + c] * di * di + b2[c];
  }
}

// ---------------- host orchestration ----------------

extern "C" void kernel_launch(void* const* d_in, const int* in_sizes, int n_in,
                              void* d_out, int out_size, void* d_ws, size_t ws_size,
                              hipStream_t stream) {
  const float* x   = (const float*)d_in[0];
  const int*   ei  = (const int*)d_in[1];
  const float* W1  = (const float*)d_in[2];
  const float* b1  = (const float*)d_in[3];
  const float* W2  = (const float*)d_in[4];
  const float* b2  = (const float*)d_in[5];
  float*       out = (float*)d_out;

  const int N = in_sizes[0] / C_IN;   // 50000 (divisible by 16)
  const int E = in_sizes[1] / 2;      // 600000
  const int* src = ei;
  const int* dst = ei + E;

  // Bump allocator over workspace (256B aligned carves), ~87 MB total.
  char*  ws  = (char*)d_ws;
  size_t off = 0;
  auto carve = [&](size_t bytes) -> void* {
    void* p = ws + off;
    off += (bytes + 255) & ~(size_t)255;
    return p;
  };

  _Float16* xh   = (_Float16*)carve((size_t)N * C_IN * sizeof(_Float16));
  _Float16* a1h  = (_Float16*)carve((size_t)N * C_HID * sizeof(_Float16));
  _Float16* W1h  = (_Float16*)carve((size_t)C_IN * C_HID * sizeof(_Float16));
  _Float16* W2h  = (_Float16*)carve((size_t)C_HID * C_OUT_PAD * sizeof(_Float16));
  float*    deg  = (float*)carve((size_t)N * sizeof(float));
  float*    dinv = (float*)carve((size_t)N * sizeof(float));
  float*    h1   = (float*)carve((size_t)N * C_HID * sizeof(float));
  float*    agg1 = (float*)carve((size_t)N * C_HID * sizeof(float));
  float*    h2   = (float*)carve((size_t)N * C_OUT_PAD * sizeof(float));

  const int T = 256;

  // Zero the accumulation targets (workspace & out are poisoned by harness).
  hipMemsetAsync(deg,  0, (size_t)N * sizeof(float), stream);
  hipMemsetAsync(agg1, 0, (size_t)N * C_HID * sizeof(float), stream);
  hipMemsetAsync(out,  0, (size_t)N * C_OUT * sizeof(float), stream);

  // f32 -> f16 operand staging.
  cvt_f32_to_f16<<<(N * C_IN + T - 1) / T, T, 0, stream>>>(x, xh, N * C_IN);
  cvt_f32_to_f16<<<(C_IN * C_HID + T - 1) / T, T, 0, stream>>>(W1, W1h, C_IN * C_HID);
  pad_w2_f16<<<(C_HID * C_OUT_PAD + T - 1) / T, T, 0, stream>>>(W2, W2h);

  // Degree (incl. self-loop via +1 in rsqrt) and D^-1/2.
  degree_kernel<<<(E + T - 1) / T, T, 0, stream>>>(dst, deg, E);
  dinv_kernel<<<(N + T - 1) / T, T, 0, stream>>>(deg, dinv, N);

  // Layer 1: h1 = x @ W1 on WMMA; then normalized edge scatter; then
  // self-loop + bias + ReLU stored as f16 for layer 2.
  wmma_gemm_k128<8><<<N / 16, 32, 0, stream>>>(xh, W1h, h1);
  scatter_edges128<<<(E * 32 + T - 1) / T, T, 0, stream>>>(src, dst, dinv, h1, agg1, E);
  selfloop_relu_f16<<<(N * C_HID + T - 1) / T, T, 0, stream>>>(agg1, h1, dinv, b1, a1h, N * C_HID);

  // Layer 2: h2 = a1 @ W2 (padded to 48 cols) on WMMA; scatter; final add.
  wmma_gemm_k128<3><<<N / 16, 32, 0, stream>>>(a1h, W2h, h2);
  scatter_edges40<<<(E * 32 + T - 1) / T, T, 0, stream>>>(src, dst, dinv, h2, out, E);
  final_add<<<(N * C_OUT + T - 1) / T, T, 0, stream>>>(h2, dinv, b2, out, N * C_OUT);
}
